// GATLayer_24154896072818
// MI455X (gfx1250) — compile-verified, matrix operand order checked by reading, not run on hardware
//
#include <hip/hip_runtime.h>
#include <hip/hip_fp16.h>

typedef __attribute__((ext_vector_type(16))) _Float16 v16h;
typedef __attribute__((ext_vector_type(8)))  _Float16 v8h;
typedef __attribute__((ext_vector_type(8)))  float    v8f;

#define N_NODES   50000
#define N_EDGES   1600000
#define IN_DIM    256
#define OUT_DIM   32
#define N_HEADS   4
#define F_DIM     128          // OUT_DIM * N_HEADS
#define ALPHA_SLOPE 0.2f
#define M_TILES   3125         // 50000 / 16
#define N_TILES   8            // 128 / 16
#define K_TILES   8            // 256 / 32

// ---------------------------------------------------------------------------
// Pack W (256x128 f32, row-major K x N) into per-fragment f16 layout so each
// lane's 16-half B fragment is one contiguous 32-byte load in the GEMM.
// Fragment layout mirrors the ISA 16-bit A/B striping:
//   lane l<16 : column n0+l,      K = kbase..kbase+7 , kbase+16..kbase+23
//   lane l>=16: column n0+(l-16), kbase = kt*32 + 8
// ---------------------------------------------------------------------------
__global__ void pack_w_kernel(const float* __restrict__ W, _Float16* __restrict__ wpack) {
    int idx = blockIdx.x * blockDim.x + threadIdx.x;   // 0 .. N_TILES*K_TILES*32-1
    if (idx >= N_TILES * K_TILES * 32) return;
    int lane = idx & 31;
    int kt   = (idx >> 5) & 7;
    int nt   = idx >> 8;
    int col  = nt * 16 + (lane & 15);
    int kbase = kt * 32 + ((lane & 16) ? 8 : 0);
    _Float16* dst = wpack + idx * 16;
    #pragma unroll
    for (int i = 0; i < 8; ++i) {
        dst[i]     = (_Float16)W[(kbase + i)      * F_DIM + col];
        dst[i + 8] = (_Float16)W[(kbase + 16 + i) * F_DIM + col];
    }
}

// ---------------------------------------------------------------------------
// Zero the output accumulator, segment-sum table, and segment-max key table.
// (harness poisons d_out/d_ws with 0xAA; we must re-init every call)
// key 0 compares <= the key of every real float (key(-inf) = 0x007FFFFF).
// ---------------------------------------------------------------------------
__global__ void init_kernel(float* __restrict__ out, float* __restrict__ ssum,
                            unsigned* __restrict__ mkey) {
    int i = blockIdx.x * blockDim.x + threadIdx.x;
    if (i < N_NODES * F_DIM)   out[i] = 0.0f;
    if (i < N_NODES * N_HEADS) { ssum[i] = 0.0f; mkey[i] = 0u; }
}

// ---------------------------------------------------------------------------
// h_prime = h @ W via v_wmma_f32_16x16x32_f16.
// Block = 256 threads = 8 waves; block owns one 16-row M tile of h (staged
// once into LDS as f16), each wave owns one 16-column N tile. K loop = 8
// WMMA ops with f32 accumulation.
// ---------------------------------------------------------------------------
__global__ void __launch_bounds__(256) gemm_kernel(const float* __restrict__ h,
                                                   const _Float16* __restrict__ wpack,
                                                   float* __restrict__ hprime) {
    __shared__ __align__(32) _Float16 tileA[16 * IN_DIM];   // 8 KB

    const int tid = threadIdx.x;
    const int m0  = blockIdx.x * 16;

    // Cooperative, coalesced load of the 16x256 f32 A tile -> f16 LDS.
    #pragma unroll
    for (int j = 0; j < 16; ++j) {
        int row = m0 + j;
        float v = (row < N_NODES) ? h[row * IN_DIM + tid] : 0.0f;
        tileA[j * IN_DIM + tid] = (_Float16)v;
    }
    __syncthreads();

    const int wave = tid >> 5;          // N tile index (0..7)
    const int lane = tid & 31;
    const int rowA = lane & 15;
    const int koff = (lane & 16) ? 8 : 0;

    v8f c = {};
    #pragma unroll
    for (int kt = 0; kt < K_TILES; ++kt) {
        const int kbase = kt * 32 + koff;
        v8h lo = *(const v8h*)&tileA[rowA * IN_DIM + kbase];        // K = kbase..+7
        v8h hi = *(const v8h*)&tileA[rowA * IN_DIM + kbase + 16];   // K = kbase+16..+23
        v16h afrag = __builtin_shufflevector(lo, hi,
            0, 1, 2, 3, 4, 5, 6, 7, 8, 9, 10, 11, 12, 13, 14, 15);
        v16h bfrag = *(const v16h*)&wpack[((wave * K_TILES + kt) * 32 + lane) * 16];
        c = __builtin_amdgcn_wmma_f32_16x16x32_f16(
                /*neg_a=*/false, afrag, /*neg_b=*/false, bfrag,
                /*c_mod=*/(short)0, c, /*reuse_a=*/false, /*reuse_b=*/false);
    }

    // C/D layout: VGPR i -> M = i (lanes 0-15) / 8+i (lanes 16-31), N = lane&15.
    const int col   = wave * 16 + (lane & 15);
    const int rbase = (lane & 16) ? 8 : 0;
    #pragma unroll
    for (int i = 0; i < 8; ++i) {
        int gr = m0 + rbase + i;
        if (gr < N_NODES) hprime[gr * F_DIM + col] = c[i];
    }
}

// ---------------------------------------------------------------------------
// Per-node attention logit halves: alpha_src[n,h] = h'[n,h,:] . a[:32,h]
//                                   alpha_dst[n,h] = h'[n,h,:] . a[32:,h]
// ---------------------------------------------------------------------------
__global__ void alpha_kernel(const float* __restrict__ hprime, const float* __restrict__ a,
                             float* __restrict__ asrc, float* __restrict__ adst) {
    int t = blockIdx.x * blockDim.x + threadIdx.x;
    if (t >= N_NODES * N_HEADS) return;
    int node = t >> 2;
    int head = t & 3;
    const float* hp = hprime + node * F_DIM + head * OUT_DIM;
    float s0 = 0.0f, s1 = 0.0f;
    #pragma unroll
    for (int d = 0; d < OUT_DIM; ++d) {
        float v = hp[d];
        s0 += v * a[d * N_HEADS + head];
        s1 += v * a[(OUT_DIM + d) * N_HEADS + head];
    }
    asrc[t] = s0;
    adst[t] = s1;
}

// Order-preserving float <-> unsigned key for atomicMax-based segment max.
__device__ __forceinline__ unsigned fkey(float f) {
    unsigned u = __float_as_uint(f);
    return (u & 0x80000000u) ? ~u : (u | 0x80000000u);
}
__device__ __forceinline__ float finvkey(unsigned k) {
    unsigned u = (k & 0x80000000u) ? (k & 0x7FFFFFFFu) : ~k;
    return __uint_as_float(u);
}

// ---------------------------------------------------------------------------
// Per-edge logits + leaky ReLU + segment max (atomic on monotone uint key).
// ---------------------------------------------------------------------------
__global__ void edge_att_kernel(const int* __restrict__ adj,
                                const float* __restrict__ asrc,
                                const float* __restrict__ adst,
                                float* __restrict__ att,
                                unsigned* __restrict__ mkey) {
    int t = blockIdx.x * blockDim.x + threadIdx.x;
    if (t >= N_EDGES * N_HEADS) return;
    int e = t >> 2, head = t & 3;
    int s = adj[e];
    int d = adj[N_EDGES + e];
    float v = asrc[s * N_HEADS + head] + adst[d * N_HEADS + head];
    v = (v > 0.0f) ? v : ALPHA_SLOPE * v;
    att[t] = v;
    atomicMax(&mkey[d * N_HEADS + head], fkey(v));
}

// ---------------------------------------------------------------------------
// e = exp(att - m[dst]); segment sum via atomicAdd.
// ---------------------------------------------------------------------------
__global__ void edge_exp_kernel(const int* __restrict__ adj,
                                float* __restrict__ att,
                                const unsigned* __restrict__ mkey,
                                float* __restrict__ ssum) {
    int t = blockIdx.x * blockDim.x + threadIdx.x;
    if (t >= N_EDGES * N_HEADS) return;
    int e = t >> 2, head = t & 3;
    int d = adj[N_EDGES + e];
    float m  = finvkey(mkey[d * N_HEADS + head]);
    float ex = __expf(att[t] - m);
    att[t] = ex;                      // overwrite in place (each t touched once)
    atomicAdd(&ssum[d * N_HEADS + head], ex);
}

// ---------------------------------------------------------------------------
// out[dst] += (e / s[dst]) * h_prime[src].  32 lanes/edge, float4 per lane,
// f32 global atomics (global_atomic_add_f32). Dominant, HBM/atomic-bound pass.
// ---------------------------------------------------------------------------
__global__ void aggregate_kernel(const int* __restrict__ adj,
                                 const float* __restrict__ att,
                                 const float* __restrict__ ssum,
                                 const float* __restrict__ hprime,
                                 float* __restrict__ out) {
    long long t = (long long)blockIdx.x * blockDim.x + threadIdx.x;
    if (t >= (long long)N_EDGES * 32) return;
    int e    = (int)(t >> 5);
    int lane = (int)(t & 31);
    int head = lane >> 3;                       // (lane*4)/32
    int s = adj[e];
    int d = adj[N_EDGES + e];
    float w = att[e * N_HEADS + head] / ssum[d * N_HEADS + head];
    const float4 hv = *(const float4*)&hprime[s * F_DIM + lane * 4];
    float* o = out + d * F_DIM + lane * 4;
    atomicAdd(o + 0, w * hv.x);
    atomicAdd(o + 1, w * hv.y);
    atomicAdd(o + 2, w * hv.z);
    atomicAdd(o + 3, w * hv.w);
}

// ---------------------------------------------------------------------------
// Workspace layout (bytes, all offsets 256B-aligned):
//   hprime : N*F*4        = 25,600,000
//   asrc   : N*NH*4       =    800,000
//   adst   : N*NH*4       =    800,000
//   att    : E*NH*4       = 25,600,000
//   mkey   : N*NH*4       =    800,000
//   ssum   : N*NH*4       =    800,000
//   wpack  : 8*8*32*16*2  =     65,536
//   total ~= 54.5 MB
// ---------------------------------------------------------------------------
extern "C" void kernel_launch(void* const* d_in, const int* in_sizes, int n_in,
                              void* d_out, int out_size, void* d_ws, size_t ws_size,
                              hipStream_t stream) {
    const float* h   = (const float*)d_in[0];
    const int*   adj = (const int*)d_in[1];
    const float* W   = (const float*)d_in[2];
    const float* a   = (const float*)d_in[3];
    float* out = (float*)d_out;

    char* ws = (char*)d_ws;
    size_t off = 0;
    float*    hprime = (float*)(ws + off);     off += (size_t)N_NODES * F_DIM * 4;
    float*    asrc   = (float*)(ws + off);     off += (size_t)N_NODES * N_HEADS * 4;
    float*    adst   = (float*)(ws + off);     off += (size_t)N_NODES * N_HEADS * 4;
    float*    att    = (float*)(ws + off);     off += (size_t)N_EDGES * N_HEADS * 4;
    unsigned* mkey   = (unsigned*)(ws + off);  off += (size_t)N_NODES * N_HEADS * 4;
    float*    ssum   = (float*)(ws + off);     off += (size_t)N_NODES * N_HEADS * 4;
    _Float16* wpack  = (_Float16*)(ws + off);  off += (size_t)N_TILES * K_TILES * 32 * 16 * 2;

    const int B = 256;

    pack_w_kernel<<<(N_TILES * K_TILES * 32 + B - 1) / B, B, 0, stream>>>(W, wpack);
    init_kernel<<<(N_NODES * F_DIM + B - 1) / B, B, 0, stream>>>(out, ssum, mkey);
    gemm_kernel<<<M_TILES, B, 0, stream>>>(h, wpack, hprime);
    alpha_kernel<<<(N_NODES * N_HEADS + B - 1) / B, B, 0, stream>>>(hprime, a, asrc, adst);
    edge_att_kernel<<<(N_EDGES * N_HEADS + B - 1) / B, B, 0, stream>>>(adj, asrc, adst, att, mkey);
    edge_exp_kernel<<<(N_EDGES * N_HEADS + B - 1) / B, B, 0, stream>>>(adj, att, mkey, ssum);
    long long aggT = (long long)N_EDGES * 32;
    aggregate_kernel<<<(unsigned)((aggT + B - 1) / B), B, 0, stream>>>(adj, att, ssum, hprime, out);
}
